// MultiHeadAttention_14980845928618
// MI455X (gfx1250) — compile-verified, matrix operand order checked by reading, not run on hardware
//
#include <hip/hip_runtime.h>
#include <hip/hip_bf16.h>

#define BDIM 2
#define SEQ 2048
#define DMODEL 1024
#define NHEAD 16
#define DHEAD 64
#define MROWS (BDIM * SEQ) /* 4096 */

typedef __attribute__((ext_vector_type(16))) _Float16 v16h;
typedef __attribute__((ext_vector_type(8)))  float    v8f;

union FragU { v16h v; unsigned u[8]; };

// Load a 16x32 f16 WMMA A/B fragment from a row-major LDS tile.
// Per CDNA5 ISA layout: lane&15 selects row; VGPR i holds K pair
// {2i,2i+1} (i<4) or {16+2(i-4),...} (i>=4); lanes 16-31 take K+8.
__device__ __forceinline__ v16h lds_load_frag(const _Float16* base, int row,
                                              int strideH, int kOff, int lane) {
  const int hi = (lane >> 4) & 1;
  const _Float16* r = base + row * strideH + kOff + hi * 8;
  FragU f;
#pragma unroll
  for (int i = 0; i < 8; ++i) {
    int k0 = (i < 4) ? (2 * i) : (16 + 2 * (i - 4));
    f.u[i] = *reinterpret_cast<const unsigned*>(r + k0);
  }
  return f.v;
}

__device__ __forceinline__ v8f wmma32(v16h a, v16h b, v8f c) {
  return __builtin_amdgcn_wmma_f32_16x16x32_f16(false, a, false, b,
                                                (short)0, c, false, false);
}

// --- CDNA5 async copy: global -> LDS via the async engine (ASYNCcnt) ---
// LDS aperture maps addr[31:0] to the LDS offset, so truncating the generic
// pointer of a __shared__ object yields the dsaddr VGPR value.
__device__ __forceinline__ void async_copy64B(const void* g, const void* l) {
  unsigned lo = (unsigned)(size_t)l;
  asm volatile(
      "global_load_async_to_lds_b128 %0, %1, off\n\t"
      "global_load_async_to_lds_b128 %0, %1, off offset:16\n\t"
      "global_load_async_to_lds_b128 %0, %1, off offset:32\n\t"
      "global_load_async_to_lds_b128 %0, %1, off offset:48"
      :: "v"(lo), "v"(g) : "memory");
}
__device__ __forceinline__ void async_copy32B(const void* g, const void* l) {
  unsigned lo = (unsigned)(size_t)l;
  asm volatile(
      "global_load_async_to_lds_b128 %0, %1, off\n\t"
      "global_load_async_to_lds_b128 %0, %1, off offset:16"
      :: "v"(lo), "v"(g) : "memory");
}
__device__ __forceinline__ void wait_async_all() {
  asm volatile("s_wait_asynccnt 0" ::: "memory");
}

__global__ void cvt_f32_to_f16_v8(const float* __restrict__ src,
                                  _Float16* __restrict__ dst, int n) {
  int i = (blockIdx.x * blockDim.x + threadIdx.x) * 8;
  if (i >= n) return;
  float4 a = *reinterpret_cast<const float4*>(src + i);
  float4 b = *reinterpret_cast<const float4*>(src + i + 4);
  union { _Float16 h[8]; uint4 u; } p;
  p.h[0] = (_Float16)a.x; p.h[1] = (_Float16)a.y;
  p.h[2] = (_Float16)a.z; p.h[3] = (_Float16)a.w;
  p.h[4] = (_Float16)b.x; p.h[5] = (_Float16)b.y;
  p.h[6] = (_Float16)b.z; p.h[7] = (_Float16)b.w;
  *reinterpret_cast<uint4*>(dst + i) = p.u;
}

// C[M=4096, N=1024] = A[M,1024] * W[N,1024]^T   (W rows are K-contiguous)
// OUT_MODE 0: f16 row-major   1: f16 scattered to Vt[b,h,dk,s]   2: f32 row-major
// K-step 64, async double-buffered LDS staging.
template <int OUT_MODE>
__global__ void __launch_bounds__(256, 1)
gemm_xt_wmma(const _Float16* __restrict__ A,
             const _Float16* __restrict__ W,
             _Float16* __restrict__ out16,
             float* __restrict__ out32, float alpha) {
  __shared__ _Float16 As[2][128][72];
  __shared__ _Float16 Ws[2][64][72];

  const int tid  = threadIdx.x;
  const int lane = tid & 31;
  const int wave = tid >> 5;                 // 0..7
  const int hi   = (lane >> 4) & 1;
  const int mBase = blockIdx.y * 128;
  const int nBase = blockIdx.x * 64;
  const int wRow  = (wave >> 1) * 32;        // 0,32,64,96
  const int wCol  = (wave & 1) * 32;         // 0,32

  v8f c00 = {}, c01 = {}, c10 = {}, c11 = {};

  const int aRow  = tid >> 1;                // 0..127
  const int aCol  = (tid & 1) * 32;          // 0 / 32
  const int wRowL = tid >> 2;                // 0..63
  const int wColL = (tid & 3) * 16;          // 0,16,32,48

  const _Float16* aG = A + (size_t)(mBase + aRow) * DMODEL + aCol;
  const _Float16* wG = W + (size_t)(nBase + wRowL) * DMODEL + wColL;

  auto stage = [&](int kk, int buf) {
    async_copy64B(aG + kk, &As[buf][aRow][aCol]);
    async_copy32B(wG + kk, &Ws[buf][wRowL][wColL]);
  };

  stage(0, 0);
  for (int kk = 0; kk < DMODEL; kk += 64) {
    const int buf = (kk >> 6) & 1;
    wait_async_all();      // current buffer landed (per-wave), then barrier
    __syncthreads();
    if (kk + 64 < DMODEL) stage(kk + 64, buf ^ 1);  // overlap next tile

    const _Float16* as  = &As[buf][0][0];
    const _Float16* wsb = &Ws[buf][0][0];
    v16h a0k0 = lds_load_frag(as, wRow + (lane & 15), 72, 0, lane);
    v16h a0k1 = lds_load_frag(as, wRow + (lane & 15), 72, 32, lane);
    v16h a1k0 = lds_load_frag(as, wRow + 16 + (lane & 15), 72, 0, lane);
    v16h a1k1 = lds_load_frag(as, wRow + 16 + (lane & 15), 72, 32, lane);
    v16h b0k0 = lds_load_frag(wsb, wCol + (lane & 15), 72, 0, lane);
    v16h b0k1 = lds_load_frag(wsb, wCol + (lane & 15), 72, 32, lane);
    v16h b1k0 = lds_load_frag(wsb, wCol + 16 + (lane & 15), 72, 0, lane);
    v16h b1k1 = lds_load_frag(wsb, wCol + 16 + (lane & 15), 72, 32, lane);

    c00 = wmma32(a0k0, b0k0, c00); c00 = wmma32(a0k1, b0k1, c00);
    c01 = wmma32(a0k0, b1k0, c01); c01 = wmma32(a0k1, b1k1, c01);
    c10 = wmma32(a1k0, b0k0, c10); c10 = wmma32(a1k1, b0k1, c10);
    c11 = wmma32(a1k0, b1k0, c11); c11 = wmma32(a1k1, b1k1, c11);
  }

#pragma unroll
  for (int t = 0; t < 4; ++t) {
    const v8f& c = (t == 0) ? c00 : (t == 1) ? c01 : (t == 2) ? c10 : c11;
    const int rOff = (t >> 1) * 16;
    const int cOff = (t & 1) * 16;
#pragma unroll
    for (int v = 0; v < 8; ++v) {
      int m = mBase + wRow + rOff + v + hi * 8;
      int n = nBase + wCol + cOff + (lane & 15);
      float val = c[v] * alpha;
      if (OUT_MODE == 2) {
        out32[(size_t)m * DMODEL + n] = val;
      } else if (OUT_MODE == 0) {
        out16[(size_t)m * DMODEL + n] = (_Float16)val;
      } else {  // Vt[b,h,dk,s]
        int bb = m >> 11, s = m & (SEQ - 1);
        int hh = n >> 6, dk = n & (DHEAD - 1);
        out16[(((size_t)bb * NHEAD + hh) * DHEAD + dk) * SEQ + s] = (_Float16)val;
      }
    }
  }
}

// Flash attention: one block per (b, h, 64-row query tile); 4 waves x 16 rows.
// K/V tiles async double-buffered. launch_bounds(...,1): occupancy is
// LDS-limited, so give the register allocator full headroom (no spills).
__global__ void __launch_bounds__(128, 1)
flash_attn_wmma(const _Float16* __restrict__ Qp,
                const _Float16* __restrict__ Kp,
                const _Float16* __restrict__ Vt,
                _Float16* __restrict__ Og) {
  __shared__ _Float16 qs[64][72];
  __shared__ _Float16 ks[2][64][72];
  __shared__ _Float16 vs[2][64][72];
  __shared__ _Float16 ps[4][16][72];

  const int tid  = threadIdx.x;   // 0..127
  const int lane = tid & 31;
  const int wave = tid >> 5;      // 0..3
  const int hi   = (lane >> 4) & 1;
  const int qb   = blockIdx.x;
  const int h    = blockIdx.y;
  const int b    = blockIdx.z;
  const int qbase = qb * 64;

  const int r  = tid >> 1;          // 0..63
  const int cH = (tid & 1) * 32;    // 0 / 32
  const _Float16* kG = Kp + (size_t)(b * SEQ + r) * DMODEL + h * DHEAD + cH;
  const _Float16* vG = Vt + ((size_t)(b * NHEAD + h) * DHEAD + r) * SEQ + cH;

  auto stage_kv = [&](int j, int buf) {
    const int kvbase = j * 64;
    async_copy64B(kG + (size_t)kvbase * DMODEL, &ks[buf][r][cH]);
    async_copy64B(vG + kvbase, &vs[buf][r][cH]);
  };

  // prologue: Q tile + first K/V tile
  async_copy64B(Qp + (size_t)(b * SEQ + qbase + r) * DMODEL + h * DHEAD + cH,
                &qs[r][cH]);
  stage_kv(0, 0);
  wait_async_all();
  __syncthreads();

  v16h aq0 = lds_load_frag(&qs[0][0], wave * 16 + (lane & 15), 72, 0, lane);
  v16h aq1 = lds_load_frag(&qs[0][0], wave * 16 + (lane & 15), 72, 32, lane);

  v8f o0 = {}, o1 = {}, o2 = {}, o3 = {};
  float mrow[8], lrow[8];
#pragma unroll
  for (int v = 0; v < 8; ++v) { mrow[v] = -__builtin_inff(); lrow[v] = 0.f; }

  for (int j = 0; j <= qb; ++j) {
    const int buf = j & 1;
    const int kvbase = j * 64;
    if (j + 1 <= qb) stage_kv(j + 1, buf ^ 1);  // overlap next tile

    // S = (q*0.125) . k^T over DK=64 (two K32 chunks), 4 column tiles.
    v8f sc[4];
#pragma unroll
    for (int t = 0; t < 4; ++t) {
      v16h bk0 = lds_load_frag(&ks[buf][0][0], t * 16 + (lane & 15), 72, 0, lane);
      v16h bk1 = lds_load_frag(&ks[buf][0][0], t * 16 + (lane & 15), 72, 32, lane);
      v8f z = {};
      z = wmma32(aq0, bk0, z);
      sc[t] = wmma32(aq1, bk1, z);
    }

    if (j == qb) {  // causal mask on the diagonal block only
#pragma unroll
      for (int t = 0; t < 4; ++t) {
        int col = kvbase + t * 16 + (lane & 15);
#pragma unroll
        for (int v = 0; v < 8; ++v) {
          int row = qbase + wave * 16 + v + hi * 8;
          if (col > row) sc[t][v] = -__builtin_inff();
        }
      }
    }

    // Online softmax; a row's 16 columns live across one 16-lane half.
    // exp() results are converted+stored straight to the LDS P tile to keep
    // register live ranges short.
#pragma unroll
    for (int v = 0; v < 8; ++v) {
      float mv = fmaxf(fmaxf(sc[0][v], sc[1][v]), fmaxf(sc[2][v], sc[3][v]));
#pragma unroll
      for (int d = 8; d >= 1; d >>= 1) mv = fmaxf(mv, __shfl_xor(mv, d, 32));
      float mnew = fmaxf(mrow[v], mv);
      float al = __expf(mrow[v] - mnew);
      float rs = 0.f;
#pragma unroll
      for (int t = 0; t < 4; ++t) {
        float p = __expf(sc[t][v] - mnew);
        ps[wave][v + hi * 8][t * 16 + (lane & 15)] = (_Float16)p;
        rs += p;
      }
#pragma unroll
      for (int d = 8; d >= 1; d >>= 1) rs += __shfl_xor(rs, d, 32);
      lrow[v] = lrow[v] * al + rs;
      mrow[v] = mnew;
      o0[v] *= al; o1[v] *= al; o2[v] *= al; o3[v] *= al;
    }
    asm volatile("s_wait_dscnt 0" ::: "memory");

    v16h pa0 = lds_load_frag(&ps[wave][0][0], (lane & 15), 72, 0, lane);
    v16h pa1 = lds_load_frag(&ps[wave][0][0], (lane & 15), 72, 32, lane);

    // O += P . V  (Vt rows are kv-contiguous => same fragment loader)
#pragma unroll
    for (int g = 0; g < 4; ++g) {
      v16h bv0 = lds_load_frag(&vs[buf][0][0], g * 16 + (lane & 15), 72, 0, lane);
      v16h bv1 = lds_load_frag(&vs[buf][0][0], g * 16 + (lane & 15), 72, 32, lane);
      v8f& o = (g == 0) ? o0 : (g == 1) ? o1 : (g == 2) ? o2 : o3;
      o = wmma32(pa0, bv0, o);
      o = wmma32(pa1, bv1, o);
    }

    if (j + 1 <= qb) {       // next buffer landed; all waves done reading cur
      wait_async_all();
      __syncthreads();
    }
  }

#pragma unroll
  for (int v = 0; v < 8; ++v) {
    float inv = 1.0f / lrow[v];
    int row = qbase + wave * 16 + v + hi * 8;
    _Float16* orow = Og + (size_t)(b * SEQ + row) * DMODEL + h * DHEAD;
    orow[0 * 16 + (lane & 15)] = (_Float16)(o0[v] * inv);
    orow[1 * 16 + (lane & 15)] = (_Float16)(o1[v] * inv);
    orow[2 * 16 + (lane & 15)] = (_Float16)(o2[v] * inv);
    orow[3 * 16 + (lane & 15)] = (_Float16)(o3[v] * inv);
  }
}

extern "C" void kernel_launch(void* const* d_in, const int* in_sizes, int n_in,
                              void* d_out, int out_size, void* d_ws, size_t ws_size,
                              hipStream_t stream) {
  (void)in_sizes; (void)n_in; (void)out_size; (void)ws_size;
  const float* Q  = (const float*)d_in[0];
  const float* K  = (const float*)d_in[1];
  const float* V  = (const float*)d_in[2];
  // d_in[3] = causal mask (recomputed analytically; unused)
  const float* wq = (const float*)d_in[4];
  const float* wk = (const float*)d_in[5];
  const float* wv = (const float*)d_in[6];
  const float* wo = (const float*)d_in[7];

  char* ws = (char*)d_ws;
  size_t off = 0;
  const size_t WBYTES = (size_t)DMODEL * DMODEL * sizeof(_Float16);  // 2 MB
  const size_t XBYTES = (size_t)MROWS * DMODEL * sizeof(_Float16);   // 8 MB
  _Float16* wq16 = (_Float16*)(ws + off); off += WBYTES;
  _Float16* wk16 = (_Float16*)(ws + off); off += WBYTES;
  _Float16* wv16 = (_Float16*)(ws + off); off += WBYTES;
  _Float16* wo16 = (_Float16*)(ws + off); off += WBYTES;
  _Float16* Q16  = (_Float16*)(ws + off); off += XBYTES;
  _Float16* K16  = (_Float16*)(ws + off); off += XBYTES;
  _Float16* V16  = (_Float16*)(ws + off); off += XBYTES;
  _Float16* Qp   = (_Float16*)(ws + off); off += XBYTES;
  _Float16* Kp   = (_Float16*)(ws + off); off += XBYTES;
  _Float16* Vt   = (_Float16*)(ws + off); off += XBYTES;
  _Float16* AO   = (_Float16*)(ws + off); off += XBYTES;

  const int nW = DMODEL * DMODEL;
  const int nX = MROWS * DMODEL;
  cvt_f32_to_f16_v8<<<(nW / 8 + 255) / 256, 256, 0, stream>>>(wq, wq16, nW);
  cvt_f32_to_f16_v8<<<(nW / 8 + 255) / 256, 256, 0, stream>>>(wk, wk16, nW);
  cvt_f32_to_f16_v8<<<(nW / 8 + 255) / 256, 256, 0, stream>>>(wv, wv16, nW);
  cvt_f32_to_f16_v8<<<(nW / 8 + 255) / 256, 256, 0, stream>>>(wo, wo16, nW);
  cvt_f32_to_f16_v8<<<(nX / 8 + 255) / 256, 256, 0, stream>>>(Q, Q16, nX);
  cvt_f32_to_f16_v8<<<(nX / 8 + 255) / 256, 256, 0, stream>>>(K, K16, nX);
  cvt_f32_to_f16_v8<<<(nX / 8 + 255) / 256, 256, 0, stream>>>(V, V16, nX);

  dim3 ggrid(DMODEL / 64, MROWS / 128);  // (16, 32)
  // fold 1/sqrt(DK)=0.125 into the Q projection
  gemm_xt_wmma<0><<<ggrid, 256, 0, stream>>>(Q16, wq16, Qp, nullptr, 0.125f);
  gemm_xt_wmma<0><<<ggrid, 256, 0, stream>>>(K16, wk16, Kp, nullptr, 1.0f);
  gemm_xt_wmma<1><<<ggrid, 256, 0, stream>>>(V16, wv16, Vt, nullptr, 1.0f);

  dim3 agrid(SEQ / 64, NHEAD, BDIM);
  flash_attn_wmma<<<agrid, 128, 0, stream>>>(Qp, Kp, Vt, AO);

  gemm_xt_wmma<2><<<ggrid, 256, 0, stream>>>(AO, wo16, nullptr, (float*)d_out, 1.0f);
}
